// ScaledDotProductAttention_39462159515702
// MI455X (gfx1250) — 8-bit matrix operand order repaired and checked by reading, compiled, not run on hardware
//
#include <hip/hip_runtime.h>
#include <stdint.h>

#define B_ 8
#define W_ 1024
#define H_ 16
#define D_ 64
#define BH_ (B_*H_)
#define NELEM ((size_t)8388608)
#define NSCORE ((size_t)134217728)

typedef int v8i __attribute__((ext_vector_type(8)));

struct QP { float scale; int zp; };

__device__ __forceinline__ unsigned fkey(float f){
  unsigned u = __float_as_uint(f);
  return (u & 0x80000000u) ? ~u : (u | 0x80000000u);
}
__device__ __forceinline__ float funkey(unsigned k){
  return __uint_as_float((k & 0x80000000u) ? (k & 0x7fffffffu) : ~k);
}
__device__ __forceinline__ unsigned umin2(unsigned a, unsigned b){ return a < b ? a : b; }
__device__ __forceinline__ unsigned umax2(unsigned a, unsigned b){ return a > b ? a : b; }
__device__ __forceinline__ int clampq(int q){ return q < -128 ? -128 : (q > 127 ? 127 : q); }

__global__ void init_mm_kernel(unsigned* mm){
  int i = threadIdx.x;
  if (i < 6){ mm[2*i] = 0xFFFFFFFFu; mm[2*i+1] = 0u; }
}

__global__ __launch_bounds__(256) void minmax_kernel(const float* __restrict__ x,
                                                     size_t n, unsigned* mm, int idx){
  __shared__ unsigned smin[8], smax[8];
  unsigned kmin = 0xFFFFFFFFu, kmax = 0u;
  size_t stride = (size_t)gridDim.x * blockDim.x;
  for (size_t i = (size_t)blockIdx.x * blockDim.x + threadIdx.x; i < n; i += stride){
    unsigned key = fkey(x[i]);
    kmin = umin2(kmin, key); kmax = umax2(kmax, key);
  }
  for (int o = 16; o; o >>= 1){
    kmin = umin2(kmin, __shfl_xor(kmin, o, 32));
    kmax = umax2(kmax, __shfl_xor(kmax, o, 32));
  }
  int wave = threadIdx.x >> 5, lane = threadIdx.x & 31;
  if (lane == 0){ smin[wave] = kmin; smax[wave] = kmax; }
  __syncthreads();
  if (threadIdx.x == 0){
    unsigned mn = smin[0], mx = smax[0];
    for (int w = 1; w < (int)(blockDim.x >> 5); ++w){ mn = umin2(mn, smin[w]); mx = umax2(mx, smax[w]); }
    atomicMin(&mm[2*idx], mn); atomicMax(&mm[2*idx+1], mx);
  }
}

__global__ void params_kernel(const unsigned* mm, QP* qp, int idx){
  float mn = funkey(mm[2*idx]), mx = funkey(mm[2*idx+1]);
  float scale = fmaxf((mx - mn) * (1.0f/255.0f), 1e-8f);
  int zp = clampq((int)rintf(-128.0f - mn/scale));
  qp[idx].scale = scale; qp[idx].zp = zp;
}

__global__ __launch_bounds__(256) void quantize_qk_kernel(const float* __restrict__ x,
                                                          int8_t* __restrict__ out,
                                                          const QP* __restrict__ qp, int pidx){
  size_t o = (size_t)blockIdx.x * blockDim.x + threadIdx.x;
  float s = qp[pidx].scale; int zp = qp[pidx].zp; float inv = 1.0f/s;
  int d = (int)(o & 63), w = (int)((o >> 6) & 1023), h = (int)((o >> 16) & 15), b = (int)(o >> 20);
  size_t in = (((size_t)b*W_ + w)*H_ + h)*D_ + d;
  out[o] = (int8_t)clampq((int)rintf(x[in]*inv) + zp);
}

__global__ __launch_bounds__(256) void quantize_vT_kernel(const float* __restrict__ x,
                                                          int8_t* __restrict__ out,
                                                          const QP* __restrict__ qp){
  size_t o = (size_t)blockIdx.x * blockDim.x + threadIdx.x;
  float s = qp[2].scale; int zp = qp[2].zp; float inv = 1.0f/s;
  int w = (int)(o & 1023), d = (int)((o >> 10) & 63), h = (int)((o >> 16) & 15), b = (int)(o >> 20);
  size_t in = (((size_t)b*W_ + w)*H_ + h)*D_ + d;
  out[o] = (int8_t)clampq((int)rintf(x[in]*inv) + zp);
}

__global__ __launch_bounds__(256) void rowsum_kernel(const int8_t* __restrict__ x,
                                                     int* __restrict__ out, int nrows, int ndw){
  int r = blockIdx.x * blockDim.x + threadIdx.x;
  if (r >= nrows) return;
  const int* p = (const int*)(x + (size_t)r * ndw * 4);
  int s = 0;
  for (int i = 0; i < ndw; ++i){
    int wv = p[i];
    s += (int)(int8_t)wv + (int)(int8_t)(wv >> 8) + (int)(int8_t)(wv >> 16) + (wv >> 24);
  }
  out[r] = s;
}

__global__ __launch_bounds__(256) void scores_wmma_kernel(
    const int8_t* __restrict__ qi8, const int8_t* __restrict__ ki8,
    const int* __restrict__ rsq_g, const int* __restrict__ rsk_g,
    const QP* __restrict__ qp, float* __restrict__ scores, unsigned* mm){
  __shared__ unsigned smin[8], smax[8];
  const int lane = threadIdx.x & 31, wave = threadIdx.x >> 5;
  const int rowtile = blockIdx.x;
  const int coltile = blockIdx.y * 8 + wave;
  const int bh = blockIdx.z;
  const int half = lane >> 4, l16 = lane & 15;
  const size_t base = (size_t)bh * W_ * D_;
  const int m = rowtile*16 + l16;
  const int n = coltile*16 + l16;
  const int* qrow = (const int*)(qi8 + base + (size_t)m * D_);
  const int* krow = (const int*)(ki8 + base + (size_t)n * D_);
  v8i a, b;
#pragma unroll
  for (int v = 0; v < 8; ++v){
    a[v] = qrow[(((v>>2)<<5) + (half<<4) + ((v&3)<<2)) >> 2];
    b[v] = krow[(((v>>2)<<5) + (half<<4) + ((v&3)<<2)) >> 2];
  }
  v8i c = {};
  c = __builtin_amdgcn_wmma_i32_16x16x64_iu8(true, a, true, b, c, false, false);

  const float fs = qp[0].scale * qp[1].scale * 0.125f;
  const int zpq = qp[0].zp, zpk = qp[1].zp;
  const int zz = D_ * zpq * zpk;
  const int* rsq = rsq_g + bh * W_;
  const int* rsk = rsk_g + bh * W_;
  float* srow = scores + (size_t)bh * W_ * W_;
  const int nc = coltile*16 + l16;
  const int ck = zpq * rsk[nc];
  unsigned kmin = 0xFFFFFFFFu, kmax = 0u;
#pragma unroll
  for (int r = 0; r < 8; ++r){
    int mc = rowtile*16 + r + (half << 3);
    float val = fs * (float)(c[r] - zpk*rsq[mc] - ck + zz);
    srow[(size_t)mc * W_ + nc] = val;
    unsigned key = fkey(val); kmin = umin2(kmin, key); kmax = umax2(kmax, key);
  }
  for (int o = 16; o; o >>= 1){ kmin = umin2(kmin, __shfl_xor(kmin, o, 32));
                                kmax = umax2(kmax, __shfl_xor(kmax, o, 32)); }
  if (lane == 0){ smin[wave] = kmin; smax[wave] = kmax; }
  __syncthreads();
  if (threadIdx.x == 0){
    unsigned mn = smin[0], mx = smax[0];
    for (int w = 1; w < 8; ++w){ mn = umin2(mn, smin[w]); mx = umax2(mx, smax[w]); }
    atomicMin(&mm[6], mn); atomicMax(&mm[7], mx);
  }
}

__global__ __launch_bounds__(256) void softmax_kernel(const float* __restrict__ scores,
                                                      const QP* __restrict__ qp,
                                                      float* __restrict__ att, unsigned* mm){
  __shared__ float redf[8];
  __shared__ unsigned redmin[8], redmax[8];
  const int row = blockIdx.x;
  const int t = threadIdx.x, lane = t & 31, wave = t >> 5;
  const float* src = scores + (size_t)row * W_;
  float* dst = att + (size_t)row * W_;
  const float ss = qp[3].scale; const int zs = qp[3].zp; const float invss = 1.0f/ss;
  float x[4]; float lmax = -3.402823466e38f;
#pragma unroll
  for (int i = 0; i < 4; ++i){
    int qv = clampq((int)rintf(src[t + 256*i] * invss) + zs);
    x[i] = (float)(qv - zs) * ss;
    lmax = fmaxf(lmax, x[i]);
  }
  for (int o = 16; o; o >>= 1) lmax = fmaxf(lmax, __shfl_xor(lmax, o, 32));
  if (lane == 0) redf[wave] = lmax;
  __syncthreads();
  float bmax = redf[0];
#pragma unroll
  for (int w = 1; w < 8; ++w) bmax = fmaxf(bmax, redf[w]);
  __syncthreads();
  float lsum = 0.f;
#pragma unroll
  for (int i = 0; i < 4; ++i){ x[i] = __expf(x[i] - bmax); lsum += x[i]; }
  for (int o = 16; o; o >>= 1) lsum += __shfl_xor(lsum, o, 32);
  if (lane == 0) redf[wave] = lsum;
  __syncthreads();
  float bsum = 0.f;
#pragma unroll
  for (int w = 0; w < 8; ++w) bsum += redf[w];
  const float inv = 1.0f / bsum;
  unsigned kmin = 0xFFFFFFFFu, kmax = 0u;
#pragma unroll
  for (int i = 0; i < 4; ++i){
    float av = x[i] * inv;
    dst[t + 256*i] = av;
    unsigned key = fkey(av); kmin = umin2(kmin, key); kmax = umax2(kmax, key);
  }
  for (int o = 16; o; o >>= 1){ kmin = umin2(kmin, __shfl_xor(kmin, o, 32));
                                kmax = umax2(kmax, __shfl_xor(kmax, o, 32)); }
  if (lane == 0){ redmin[wave] = kmin; redmax[wave] = kmax; }
  __syncthreads();
  if (t == 0){
    unsigned mn = redmin[0], mx = redmax[0];
    for (int w = 1; w < 8; ++w){ mn = umin2(mn, redmin[w]); mx = umax2(mx, redmax[w]); }
    atomicMin(&mm[8], mn); atomicMax(&mm[9], mx);
  }
}

__global__ __launch_bounds__(256) void quant_att_kernel(float* __restrict__ att,
                                                        int8_t* __restrict__ atti8,
                                                        const QP* __restrict__ qp){
  size_t i = (size_t)blockIdx.x * blockDim.x + threadIdx.x;
  const float s = qp[4].scale; const int zp = qp[4].zp; const float inv = 1.0f/s;
  int qv = clampq((int)rintf(att[i]*inv) + zp);
  atti8[i] = (int8_t)qv;
  att[i] = (float)(qv - zp) * s;
}

__global__ __launch_bounds__(128) void context_wmma_kernel(
    const int8_t* __restrict__ atti8, const int8_t* __restrict__ vti8,
    const int* __restrict__ rsa_g, const int* __restrict__ csv_g,
    const QP* __restrict__ qp, float* __restrict__ ctx, unsigned* mm){
  __shared__ unsigned smin[4], smax[4];
  const int lane = threadIdx.x & 31, wave = threadIdx.x >> 5;
  const int rowtile = blockIdx.x, bh = blockIdx.y, dtile = wave;
  const int half = lane >> 4, l16 = lane & 15;
  const int m = rowtile*16 + l16;
  const int n = dtile*16 + l16;
  const int* arow = (const int*)(atti8 + (size_t)bh*W_*W_ + (size_t)m*W_);
  const int* vrow = (const int*)(vti8  + (size_t)bh*D_*W_ + (size_t)n*W_);
  v8i acc = {};
  for (int kb = 0; kb < W_/64; ++kb){
    v8i a, b;
#pragma unroll
    for (int v = 0; v < 8; ++v){
      a[v] = arow[(kb*64 + ((v>>2)<<5) + (half<<4) + ((v&3)<<2)) >> 2];
      b[v] = vrow[(kb*64 + ((v>>2)<<5) + (half<<4) + ((v&3)<<2)) >> 2];
    }
    acc = __builtin_amdgcn_wmma_i32_16x16x64_iu8(true, a, true, b, acc, false, false);
  }
  const float fs = qp[4].scale * qp[2].scale;
  const int za = qp[4].zp, zv = qp[2].zp;
  const int zz = W_ * za * zv;
  const int* rsa = rsa_g + bh * W_;
  const int* csv = csv_g + bh * D_;
  float* cbase = ctx + (size_t)bh * W_ * D_;
  const int nc = dtile*16 + l16;
  const int cv = za * csv[nc];
  unsigned kmin = 0xFFFFFFFFu, kmax = 0u;
#pragma unroll
  for (int r = 0; r < 8; ++r){
    int mc = rowtile*16 + r + (half << 3);
    float val = fs * (float)(acc[r] - cv - zv*rsa[mc] + zz);
    cbase[(size_t)mc * D_ + nc] = val;
    unsigned key = fkey(val); kmin = umin2(kmin, key); kmax = umax2(kmax, key);
  }
  for (int o = 16; o; o >>= 1){ kmin = umin2(kmin, __shfl_xor(kmin, o, 32));
                                kmax = umax2(kmax, __shfl_xor(kmax, o, 32)); }
  if (lane == 0){ smin[wave] = kmin; smax[wave] = kmax; }
  __syncthreads();
  if (threadIdx.x == 0){
    unsigned mn = smin[0], mx = smax[0];
    for (int w = 1; w < 4; ++w){ mn = umin2(mn, smin[w]); mx = umax2(mx, smax[w]); }
    atomicMin(&mm[10], mn); atomicMax(&mm[11], mx);
  }
}

__global__ __launch_bounds__(256) void quant_ctx_kernel(const float* __restrict__ ctx,
                                                        float* __restrict__ out,
                                                        const QP* __restrict__ qp){
  size_t o = (size_t)blockIdx.x * blockDim.x + threadIdx.x;
  const float s = qp[5].scale; const int zp = qp[5].zp; const float inv = 1.0f/s;
  int d = (int)(o & 63), h = (int)((o >> 6) & 15), w = (int)((o >> 10) & 1023), b = (int)(o >> 20);
  size_t ci = (((size_t)b*H_ + h)*W_ + w)*D_ + d;
  int qv = clampq((int)rintf(ctx[ci]*inv) + zp);
  out[o] = (float)(qv - zp) * s;
}

extern "C" void kernel_launch(void* const* d_in, const int* in_sizes, int n_in,
                              void* d_out, int out_size, void* d_ws, size_t ws_size,
                              hipStream_t stream){
  (void)in_sizes; (void)n_in; (void)out_size; (void)ws_size;
  const float* q = (const float*)d_in[0];
  const float* k = (const float*)d_in[1];
  const float* v = (const float*)d_in[2];
  float* ctx_out = (float*)d_out;
  float* att_out = (float*)d_out + NELEM;

  char* ws = (char*)d_ws;
  size_t off = 0;
  int8_t* qi8   = (int8_t*)(ws + off); off += NELEM;
  int8_t* ki8   = (int8_t*)(ws + off); off += NELEM;
  int8_t* vti8  = (int8_t*)(ws + off); off += NELEM;
  int8_t* atti8 = (int8_t*)(ws + off); off += NSCORE;
  float*  scores= (float*)(ws + off);  off += NSCORE * sizeof(float);
  float*  ctxf  = (float*)(ws + off);  off += NELEM * sizeof(float);
  int* rsq = (int*)(ws + off); off += (size_t)BH_ * W_ * 4;
  int* rsk = (int*)(ws + off); off += (size_t)BH_ * W_ * 4;
  int* rsa = (int*)(ws + off); off += (size_t)BH_ * W_ * 4;
  int* csv = (int*)(ws + off); off += (size_t)BH_ * D_ * 4;
  unsigned* mm = (unsigned*)(ws + off); off += 256;
  QP* qp = (QP*)(ws + off); off += 256;

  init_mm_kernel<<<1, 32, 0, stream>>>(mm);
  minmax_kernel<<<4096, 256, 0, stream>>>(q, NELEM, mm, 0);
  minmax_kernel<<<4096, 256, 0, stream>>>(k, NELEM, mm, 1);
  minmax_kernel<<<4096, 256, 0, stream>>>(v, NELEM, mm, 2);
  params_kernel<<<1, 1, 0, stream>>>(mm, qp, 0);
  params_kernel<<<1, 1, 0, stream>>>(mm, qp, 1);
  params_kernel<<<1, 1, 0, stream>>>(mm, qp, 2);
  quantize_qk_kernel<<<(unsigned)(NELEM/256), 256, 0, stream>>>(q, qi8, qp, 0);
  quantize_qk_kernel<<<(unsigned)(NELEM/256), 256, 0, stream>>>(k, ki8, qp, 1);
  quantize_vT_kernel<<<(unsigned)(NELEM/256), 256, 0, stream>>>(v, vti8, qp);
  rowsum_kernel<<<(BH_*W_)/256, 256, 0, stream>>>(qi8, rsq, BH_*W_, D_/4);
  rowsum_kernel<<<(BH_*W_)/256, 256, 0, stream>>>(ki8, rsk, BH_*W_, D_/4);
  rowsum_kernel<<<(BH_*D_)/256, 256, 0, stream>>>(vti8, csv, BH_*D_, W_/4);
  scores_wmma_kernel<<<dim3(W_/16, W_/128, BH_), 256, 0, stream>>>(qi8, ki8, rsq, rsk, qp, scores, mm);
  params_kernel<<<1, 1, 0, stream>>>(mm, qp, 3);
  softmax_kernel<<<BH_*W_, 256, 0, stream>>>(scores, qp, att_out, mm);
  params_kernel<<<1, 1, 0, stream>>>(mm, qp, 4);
  quant_att_kernel<<<(unsigned)(NSCORE/256), 256, 0, stream>>>(att_out, atti8, qp);
  rowsum_kernel<<<(BH_*W_)/256, 256, 0, stream>>>(atti8, rsa, BH_*W_, W_/4);
  context_wmma_kernel<<<dim3(W_/16, BH_), 128, 0, stream>>>(atti8, vti8, rsa, csv, qp, ctxf, mm);
  params_kernel<<<1, 1, 0, stream>>>(mm, qp, 5);
  quant_ctx_kernel<<<(unsigned)(NELEM/256), 256, 0, stream>>>(ctxf, ctx_out, qp);
}
